// ClusterAttn_15109694947341
// MI455X (gfx1250) — matrix operand order repaired and checked by reading, not compiled, not run on hardware
//
#include <hip/hip_runtime.h>
#include <hip/hip_bf16.h>
#include <math.h>

// ---------------------------------------------------------------------------
// Types for CDNA5 WMMA (wave32): v_wmma_f32_16x16x32_bf16
// ---------------------------------------------------------------------------
typedef __bf16 bf16_t;
typedef __attribute__((ext_vector_type(8)))  __bf16 bf16x8;   // 16B, natural align 16
typedef __attribute__((ext_vector_type(16))) __bf16 bf16x16;  // register fragment
typedef __attribute__((ext_vector_type(8)))  float  floatx8;
typedef __attribute__((ext_vector_type(4)))  unsigned int u32x4;
typedef __attribute__((ext_vector_type(8)))  int    i32x8;
typedef __attribute__((ext_vector_type(4)))  int    i32x4;

#if defined(__HIP_DEVICE_COMPILE__) && defined(__gfx1250__) && \
    __has_builtin(__builtin_amdgcn_tensor_load_to_lds)
#define HAVE_TDM 1
#else
#define HAVE_TDM 0
#endif

__device__ __forceinline__ bf16_t to_bf16(float x)  { return (bf16_t)x; }
__device__ __forceinline__ bf16_t to_bf16(bf16_t x) { return x; }

// Build a 16-elem fragment from two contiguous 8-elem (16B) chunks.
__device__ __forceinline__ bf16x16 frag16_split(const bf16_t* plo, const bf16_t* phi) {
    bf16x8 lo = *(const bf16x8*)plo;
    bf16x8 hi = *(const bf16x8*)phi;
    return __builtin_shufflevector(lo, hi, 0,1,2,3,4,5,6,7,8,9,10,11,12,13,14,15);
}
// Fragment from one contiguous 16-elem run (two 16B chunks).
__device__ __forceinline__ bf16x16 frag16(const bf16_t* p) {
    return frag16_split(p, p + 8);
}

#if HAVE_TDM
// Tensor Data Mover: 2D tile load (bf16) global -> LDS, tracked by TENSORcnt.
// D# layout per CDNA5 ISA ch.8 (group0: control/addresses, group1: dims).
__device__ __forceinline__ void tdm_load_2d_bf16(unsigned lds_off,
                                                 unsigned long long gaddr,
                                                 int width, int height, int stride) {
    u32x4 g0;
    g0[0] = 1u;                                            // count=1, user mode
    g0[1] = lds_off;                                       // lds_addr [63:32]
    g0[2] = (unsigned)(gaddr & 0xffffffffull);             // global_addr lo
    g0[3] = (unsigned)((gaddr >> 32) & 0x01ffffffull)      // global_addr hi (57b)
          | (2u << 30);                                    // type=2 ("image")
    i32x8 g1;
    g1[0] = (int)(1u << 16);                               // data_size=1 -> 2 bytes
    g1[1] = (int)(((unsigned)width & 0xffffu) << 16);      // tensor_dim0 [63:48]
    g1[2] = (int)((((unsigned)width >> 16) & 0xffffu)      // tensor_dim0 [79:64]
          | (((unsigned)height & 0xffffu) << 16));         // tensor_dim1 [95:80]
    g1[3] = (int)((((unsigned)height >> 16) & 0xffffu)     // tensor_dim1 [111:96]
          | (((unsigned)width & 0xffffu) << 16));          // tile_dim0  [127:112]
    g1[4] = (int)((unsigned)height & 0xffffu);             // tile_dim1  [143:128]
    g1[5] = stride;                                        // dim0_stride lo32
    g1[6] = 0;
    g1[7] = 0;
    i32x4 z4 = {0, 0, 0, 0};
#if __clang_major__ >= 23
    i32x8 z8 = {0, 0, 0, 0, 0, 0, 0, 0};
    __builtin_amdgcn_tensor_load_to_lds(g0, g1, z4, z4, z8, 0);
#else
    __builtin_amdgcn_tensor_load_to_lds(g0, g1, z4, z4, 0);
#endif
}
#endif

// A-matrix (16x32): lane (half,m) takes k in [16*half, 16*half+16) of its row in this code (same order as B).
// B-matrix (32x16): lane (half,n) takes k in [16*half, 16*half+16) of its column in this code (same order as A).

// ---------------------------------------------------------------------------
// f32 -> bf16 conversion (weights)
// ---------------------------------------------------------------------------
__global__ void cvt_f32_bf16_kernel(const float* __restrict__ src,
                                    bf16_t* __restrict__ dst, int n) {
    int i = blockIdx.x * 256 + threadIdx.x;
    if (i < n) dst[i] = (bf16_t)src[i];
}

// ---------------------------------------------------------------------------
// Tiled WMMA GEMM:  C[M,N] = A[M,K] @ B[K,N] (+ bias[N])
// Block = 256 threads (8 waves). Tile 64(M) x 128(N), K step 32.
// A tile: row-major LDS [64][40]. B tile: n-major LDS [128][40].
// ---------------------------------------------------------------------------
template <typename TA, typename TOUT>
__global__ __launch_bounds__(256)
void gemm_wmma_kernel(const TA* __restrict__ A, const bf16_t* __restrict__ B,
                      const float* __restrict__ bias, TOUT* __restrict__ C,
                      int M, int N, int K) {
    __shared__ bf16_t As[64][40];    // m-major, natural k order
    __shared__ bf16_t Bn[128][40];   // n-major (transposed), natural k order

    const int tid  = threadIdx.x;
    const int lane = tid & 31;
    const int wave = tid >> 5;
    const int half = lane >> 4;
    const int ml   = lane & 15;
    const int bm   = blockIdx.x * 64;
    const int bn   = blockIdx.y * 128;
    const int mt   = wave & 3;   // m tile
    const int nh   = wave >> 2;  // n half (0/1): cols nh*64 + j*16

    floatx8 acc[4];
    #pragma unroll
    for (int j = 0; j < 4; ++j)
        #pragma unroll
        for (int i = 0; i < 8; ++i) acc[j][i] = 0.f;

    const int ar  = tid >> 2;        // 0..63  (A row)
    const int ac  = (tid & 3) * 8;   // 0..24  (A k chunk)
    const int bnl = tid & 127;       // B n lane
    const int bkg = tid >> 7;        // 0..1   (B k-group base)

    for (int k0 = 0; k0 < K; k0 += 32) {
        // Stage A tile (64x32): 8 contiguous k per thread -> one b128 store.
        {
            const TA* src = A + (size_t)(bm + ar) * K + k0 + ac;
            bf16x8 t;
            #pragma unroll
            for (int e = 0; e < 8; ++e) t[e] = to_bf16(src[e]);
            *(bf16x8*)&As[ar][ac] = t;
        }
        // Stage B tile (32x128) n-major: 8 k-strided loads (coalesced across
        // lanes, consecutive n), one contiguous b128 store per group.
        #pragma unroll
        for (int s = 0; s < 2; ++s) {
            const int kg = bkg + 2 * s;  // 0..3
            const bf16_t* src = B + (size_t)(k0 + kg * 8) * N + bn + bnl;
            bf16x8 t;
            #pragma unroll
            for (int e = 0; e < 8; ++e) t[e] = src[(size_t)e * N];
            *(bf16x8*)&Bn[bnl][kg * 8] = t;
        }
        if (k0 + 32 < K) {
            __builtin_prefetch(A + (size_t)(bm + ar) * K + k0 + 32 + ac, 0, 3);
            __builtin_prefetch(B + (size_t)(k0 + 32) * N + bn + bnl, 0, 3);
        }
        __syncthreads();

        const bf16_t* arow = &As[mt * 16 + ml][0];
        bf16x16 af = frag16(arow + 16 * half);
        #pragma unroll
        for (int j = 0; j < 4; ++j) {
            bf16x16 bfj = frag16(&Bn[nh * 64 + j * 16 + ml][16 * half]);
            acc[j] = __builtin_amdgcn_wmma_f32_16x16x32_bf16(false, af, false, bfj,
                                                             (short)0, acc[j], false, false);
        }
        __syncthreads();
    }

    #pragma unroll
    for (int j = 0; j < 4; ++j) {
        const int col = bn + nh * 64 + j * 16 + ml;
        const float bv = bias ? bias[col] : 0.f;
        #pragma unroll
        for (int i = 0; i < 8; ++i) {
            const int row = bm + mt * 16 + i + half * 8;
            C[(size_t)row * N + col] = (TOUT)(acc[j][i] + bv);
        }
    }
}

// ---------------------------------------------------------------------------
// Gate: ga = sigmoid(fea @ W_ga + b_ga), fea bf16 (32768 x 1536), W_ga f32.
// ---------------------------------------------------------------------------
__global__ __launch_bounds__(256)
void ga_kernel(const bf16_t* __restrict__ fea, const float* __restrict__ Wga,
               const float* __restrict__ bga, float* __restrict__ ga) {
    const int t   = blockIdx.x * 256 + threadIdx.x;   // 262144
    const int row = t >> 3;
    const int g   = t & 7;
    const bf16x8* fr = (const bf16x8*)(fea + (size_t)row * 1536);
    float s = bga[g];
    for (int k8 = 0; k8 < 192; ++k8) {
        bf16x8 v = fr[k8];
        #pragma unroll
        for (int e = 0; e < 8; ++e) s += (float)v[e] * Wga[(k8 * 8 + e) * 8 + g];
    }
    ga[t] = 1.f / (1.f + __expf(-s));
}

// ---------------------------------------------------------------------------
// BN1 + softmax(C=64) + gate multiply. One thread per (row,g) chunk of 64.
// ---------------------------------------------------------------------------
__global__ __launch_bounds__(256)
void act_softmax_kernel(const float* __restrict__ act,
                        const float* __restrict__ g1, const float* __restrict__ b1,
                        const float* __restrict__ m1, const float* __restrict__ v1,
                        const float* __restrict__ ga, bf16_t* __restrict__ outp) {
    const int t  = blockIdx.x * 256 + threadIdx.x;  // 262144
    const int g  = t & 7;
    const int jb = g * 64;
    const float* a = act + (size_t)t * 64;          // t*64 == row*512 + g*64
    float vbuf[64];
    float mx = -1e30f;
    #pragma unroll 4
    for (int c = 0; c < 64; ++c) {
        const float sc = g1[jb + c] * rsqrtf(v1[jb + c] + 1e-5f);
        const float x  = (a[c] - m1[jb + c]) * sc + b1[jb + c];
        vbuf[c] = x;
        mx = fmaxf(mx, x);
    }
    float s = 0.f;
    #pragma unroll 4
    for (int c = 0; c < 64; ++c) { vbuf[c] = __expf(vbuf[c] - mx); s += vbuf[c]; }
    const float w = ga[t] / s;
    bf16_t* o = outp + (size_t)t * 64;
    #pragma unroll 4
    for (int c = 0; c < 64; ++c) o[c] = (bf16_t)(vbuf[c] * w);
}

// ---------------------------------------------------------------------------
// cent[b,c,f] = sum_n act[b,n,c] * fea_g[b,n,f]   (A^T . B per batch)
// ---------------------------------------------------------------------------
__global__ __launch_bounds__(256)
void cent_wmma_kernel(const bf16_t* __restrict__ act,
                      const bf16_t* __restrict__ fea,
                      float* __restrict__ cent) {
    __shared__ bf16_t AT[64][40];    // c-major: AT[c][n_local]
    __shared__ bf16_t FB[192][40];   // f-major: FB[f][n_local]

    const int b    = blockIdx.x;
    const int tid  = threadIdx.x;
    const int lane = tid & 31;
    const int wave = tid >> 5;
    const int half = lane >> 4;
    const int ml   = lane & 15;
    const int mt   = wave & 3;       // c tile: c0 = mt*16
    const int ng   = wave >> 2;      // 0/1 -> f offset ng*96

    floatx8 acc[6];
    #pragma unroll
    for (int j = 0; j < 6; ++j)
        #pragma unroll
        for (int i = 0; i < 8; ++i) acc[j][i] = 0.f;

    const size_t base_a = (size_t)b * 8192 * 64;
    const size_t base_f = (size_t)b * 8192 * 192;
    const int acl = tid & 63;        // act c lane
    const int akg = tid >> 6;        // 0..3 act k-group

    for (int n0 = 0; n0 < 8192; n0 += 32) {
        {
            const bf16_t* src = act + base_a + (size_t)(n0 + akg * 8) * 64 + acl;
            bf16x8 t;
            #pragma unroll
            for (int e = 0; e < 8; ++e) t[e] = src[(size_t)e * 64];
            *(bf16x8*)&AT[acl][akg * 8] = t;
        }
        #pragma unroll
        for (int s = 0; s < 3; ++s) {
            const int slot = s * 256 + tid;
            const int f  = slot % 192;
            const int kg = slot / 192;   // 0..3
            const bf16_t* src = fea + base_f + (size_t)(n0 + kg * 8) * 192 + f;
            bf16x8 t;
            #pragma unroll
            for (int e = 0; e < 8; ++e) t[e] = src[(size_t)e * 192];
            *(bf16x8*)&FB[f][kg * 8] = t;
        }
        __syncthreads();

        const bf16_t* arow = &AT[mt * 16 + ml][0];
        bf16x16 af = frag16(arow + 16 * half);
        #pragma unroll
        for (int j = 0; j < 6; ++j) {
            bf16x16 bfj = frag16(&FB[ng * 96 + j * 16 + ml][16 * half]);
            acc[j] = __builtin_amdgcn_wmma_f32_16x16x32_bf16(false, af, false, bfj,
                                                             (short)0, acc[j], false, false);
        }
        __syncthreads();
    }

    #pragma unroll
    for (int j = 0; j < 6; ++j) {
        const int f = ng * 96 + j * 16 + ml;
        #pragma unroll
        for (int i = 0; i < 8; ++i) {
            const int c = mt * 16 + i + 8 * half;
            cent[((size_t)b * 64 + c) * 192 + f] = acc[j][i];
        }
    }
}

// ---------------------------------------------------------------------------
// BN2: nc f32 (2048 x 768), channel c = row%64, out bf16.
// ---------------------------------------------------------------------------
__global__ __launch_bounds__(256)
void bn2_kernel(const float* __restrict__ nc,
                const float* __restrict__ g2, const float* __restrict__ b2,
                const float* __restrict__ m2, const float* __restrict__ v2,
                bf16_t* __restrict__ outp, int n) {
    const int i = blockIdx.x * 256 + threadIdx.x;
    if (i >= n) return;
    const int row = i / 768;
    const int c   = row & 63;
    const float sc = g2[c] * rsqrtf(v2[c] + 1e-5f);
    outp[i] = (bf16_t)((nc[i] - m2[c]) * sc + b2[c]);
}

// ---------------------------------------------------------------------------
// Fused attention. Block = (batch, 128-row chunk of S), 8 waves x 16 rows.
// KV tile (64x768 bf16) is DMA'd by the Tensor Data Mover into a packed LDS
// region (TENSORcnt-tracked), then redistributed into padded Kt / transposed
// VtT for conflict-free b128 fragment loads. Falls back to vector global
// staging if the TDM builtin is unavailable.
// ---------------------------------------------------------------------------
#define ATTN_RAW_W   768
#define ATTN_KSTRIDE 408
#define ATTN_VSTRIDE 40
#define ATTN_ASTRIDE 40
#define ATTN_SMEM_ELEMS (64*ATTN_RAW_W + 64*ATTN_KSTRIDE + 384*ATTN_VSTRIDE + 8*16*ATTN_ASTRIDE)
#define ATTN_SMEM_BYTES (ATTN_SMEM_ELEMS * (int)sizeof(bf16_t))

__global__ __launch_bounds__(256)
void attn_kernel(const bf16_t* __restrict__ q, const bf16_t* __restrict__ kv,
                 bf16_t* __restrict__ outp) {
    extern __shared__ bf16_t smem[];
    bf16_t (*RAW)[ATTN_RAW_W]   = (bf16_t(*)[ATTN_RAW_W])smem;
    bf16_t (*Kt)[ATTN_KSTRIDE]  = (bf16_t(*)[ATTN_KSTRIDE])(smem + 64 * ATTN_RAW_W);
    bf16_t (*VtT)[ATTN_VSTRIDE] =
        (bf16_t(*)[ATTN_VSTRIDE])(smem + 64 * ATTN_RAW_W + 64 * ATTN_KSTRIDE);
    bf16_t* attnBase = smem + 64 * ATTN_RAW_W + 64 * ATTN_KSTRIDE + 384 * ATTN_VSTRIDE;

    const int b     = blockIdx.x >> 3;
    const int chunk = blockIdx.x & 7;
    const int tid   = threadIdx.x;
    const int lane  = tid & 31;
    const int wave  = tid >> 5;
    const int half  = lane >> 4;
    const int ml    = lane & 15;

#if HAVE_TDM
    // One TDM descriptor moves the whole 64x768 bf16 KV tile into LDS.
    if (wave == 0) {
        tdm_load_2d_bf16((unsigned)(uintptr_t)&RAW[0][0],
                         (unsigned long long)(uintptr_t)(kv + (size_t)b * 64 * 768),
                         768, 64, 768);
        __builtin_amdgcn_s_wait_tensorcnt(0);
    }
    __syncthreads();
    // Redistribute LDS->LDS: Kt padded rows (b128 copy), VtT transposed.
    #pragma unroll
    for (int s = 0; s < 12; ++s) {
        const int slot = s * 256 + tid;          // 0..3071
        const int c  = slot / 48;
        const int cg = slot % 48;
        *(bf16x8*)&Kt[c][cg * 8] = *(const bf16x8*)&RAW[c][cg * 8];
    }
    #pragma unroll
    for (int s = 0; s < 12; ++s) {
        const int slot = s * 256 + tid;          // 0..3071
        const int p  = slot % 384;
        const int cg = slot / 384;               // 0..7
        bf16x8 t;
        #pragma unroll
        for (int e = 0; e < 8; ++e) t[e] = RAW[cg * 8 + e][384 + p];
        *(bf16x8*)&VtT[p][cg * 8] = t;
    }
#else
    // Fallback: direct vector staging from global.
    #pragma unroll
    for (int s = 0; s < 12; ++s) {
        const int slot = s * 256 + tid;
        const int c  = slot / 48;
        const int cg = slot % 48;
        bf16x8 t = *(const bf16x8*)(kv + ((size_t)b * 64 + c) * 768 + cg * 8);
        *(bf16x8*)&Kt[c][cg * 8] = t;
    }
    #pragma unroll
    for (int s = 0; s < 12; ++s) {
        const int slot = s * 256 + tid;
        const int p  = slot % 384;
        const int cg = slot / 384;
        const bf16_t* src = kv + ((size_t)b * 64 + cg * 8) * 768 + 384 + p;
        bf16x8 t;
        #pragma unroll
        for (int e = 0; e < 8; ++e) t[e] = src[(size_t)e * 768];
        *(bf16x8*)&VtT[p][cg * 8] = t;
    }
#endif
    __syncthreads();

    const int srow0 = b * 1024 + chunk * 128 + wave * 16;

    // scores = q . K^T  (16 rows x 64 cols per wave)
    floatx8 sacc[4];
    #pragma unroll
    for (int j = 0; j < 4; ++j)
        #pragma unroll
        for (int i = 0; i < 8; ++i) sacc[j][i] = 0.f;

    for (int k0 = 0; k0 < 384; k0 += 32) {
        const bf16_t* qr = q + (size_t)(srow0 + ml) * 384 + k0;
        bf16x16 af = frag16(qr + 16 * half);
        #pragma unroll
        for (int j = 0; j < 4; ++j) {
            bf16x16 bfj = frag16(&Kt[j * 16 + ml][k0 + 16 * half]);
            sacc[j] = __builtin_amdgcn_wmma_f32_16x16x32_bf16(false, af, false, bfj,
                                                              (short)0, sacc[j], false, false);
        }
    }

    // softmax over 64 cols; reductions stay inside 16-lane halves (C/D layout).
    const float rs = 0.051031036307982884f;  // 1/sqrt(384)
    bf16_t (*attnS)[ATTN_ASTRIDE] =
        (bf16_t(*)[ATTN_ASTRIDE])(attnBase + wave * 16 * ATTN_ASTRIDE);
    #pragma unroll
    for (int i = 0; i < 8; ++i) {
        float vj[4], mx = -1e30f;
        #pragma unroll
        for (int j = 0; j < 4; ++j) { vj[j] = sacc[j][i] * rs; mx = fmaxf(mx, vj[j]); }
        #pragma unroll
        for (int d = 8; d >= 1; d >>= 1) mx = fmaxf(mx, __shfl_xor(mx, d, 32));
        float pj[4], s = 0.f;
        #pragma unroll
        for (int j = 0; j < 4; ++j) { pj[j] = __expf(vj[j] - mx); s += pj[j]; }
        #pragma unroll
        for (int d = 8; d >= 1; d >>= 1) s += __shfl_xor(s, d, 32);
        const float inv = 1.f / s;
        const int r = i + 8 * half;
        #pragma unroll
        for (int j = 0; j < 4; ++j) attnS[r][j * 16 + ml] = (bf16_t)(pj[j] * inv);
    }
    // Same-wave LDS ops are in-order; no barrier needed before re-reading attnS.

    // out = attn (16x64) @ V (64x384)
    for (int p0 = 0; p0 < 384; p0 += 16) {
        floatx8 oacc;
        #pragma unroll
        for (int i = 0; i < 8; ++i) oacc[i] = 0.f;
        #pragma unroll
        for (int kk = 0; kk < 2; ++kk) {
            const bf16_t* arow = &attnS[ml][kk * 32];
            bf16x16 af  = frag16(arow + 16 * half);
            bf16x16 bfv = frag16(&VtT[p0 + ml][kk * 32 + 16 * half]);
            oacc = __builtin_amdgcn_wmma_f32_16x16x32_bf16(false, af, false, bfv,
                                                           (short)0, oacc, false, false);
        }
        #pragma unroll
        for (int i = 0; i < 8; ++i)
            outp[(size_t)(srow0 + i + 8 * half) * 384 + p0 + ml] = (bf16_t)oacc[i];
    }
}

// ---------------------------------------------------------------------------
// Host launch
// ---------------------------------------------------------------------------
static inline size_t align_up(size_t x, size_t a) { return (x + a - 1) & ~(a - 1); }

extern "C" void kernel_launch(void* const* d_in, const int* in_sizes, int n_in,
                              void* d_out, int out_size, void* d_ws, size_t ws_size,
                              hipStream_t stream) {
    (void)in_sizes; (void)n_in; (void)out_size; (void)ws_size;
    const float* x       = (const float*)d_in[0];
    const float* W_exp   = (const float*)d_in[1];
    const float* b_exp   = (const float*)d_in[2];
    const float* W_ga    = (const float*)d_in[3];
    const float* b_ga    = (const float*)d_in[4];
    const float* bn1_g   = (const float*)d_in[5];
    const float* bn1_b   = (const float*)d_in[6];
    const float* bn1_m   = (const float*)d_in[7];
    const float* bn1_v   = (const float*)d_in[8];
    const float* cw      = (const float*)d_in[9];
    const float* W_proj  = (const float*)d_in[10];
    const float* b_proj  = (const float*)d_in[11];
    const float* bn2_g   = (const float*)d_in[12];
    const float* bn2_b   = (const float*)d_in[13];
    const float* bn2_m   = (const float*)d_in[14];
    const float* bn2_v   = (const float*)d_in[15];
    const float* Wq      = (const float*)d_in[16];
    const float* bq      = (const float*)d_in[17];
    const float* Wkv     = (const float*)d_in[18];
    const float* bkv     = (const float*)d_in[19];
    const float* Wp2     = (const float*)d_in[20];
    const float* bp2     = (const float*)d_in[21];
    float* out = (float*)d_out;

    const int M  = 32768;           // B*S
    const int BC = 2048;            // B*C

    char* ws = (char*)d_ws;
    size_t off = 0;
    auto carve = [&](size_t bytes) { void* p = ws + off; off = align_up(off + bytes, 256); return p; };

    bf16_t* fea_bf   = (bf16_t*)carve((size_t)M * 1536 * 2);
    float*  act_f    = (float* )carve((size_t)M * 512 * 4);
    bf16_t* act_soft = (bf16_t*)carve((size_t)M * 512 * 2);
    float*  ga       = (float* )carve((size_t)M * 8 * 4);
    float*  cent     = (float* )carve((size_t)BC * 192 * 4);
    float*  nc_f     = (float* )carve((size_t)BC * 768 * 4);
    bf16_t* nc_bf    = (bf16_t*)carve((size_t)BC * 768 * 2);
    bf16_t* q_bf     = (bf16_t*)carve((size_t)M * 384 * 2);
    bf16_t* kv_bf    = (bf16_t*)carve((size_t)BC * 768 * 2);
    bf16_t* ao_bf    = (bf16_t*)carve((size_t)M * 384 * 2);
    bf16_t* wexp_bf  = (bf16_t*)carve((size_t)768 * 1536 * 2);
    bf16_t* cw_bf    = (bf16_t*)carve((size_t)1536 * 512 * 2);
    bf16_t* wproj_bf = (bf16_t*)carve((size_t)192 * 768 * 2);
    bf16_t* wq_bf    = (bf16_t*)carve((size_t)768 * 384 * 2);
    bf16_t* wkv_bf   = (bf16_t*)carve((size_t)768 * 768 * 2);
    bf16_t* wp2_bf   = (bf16_t*)carve((size_t)384 * 768 * 2);

    auto cvt = [&](const float* s, bf16_t* d, int n) {
        cvt_f32_bf16_kernel<<<(n + 255) / 256, 256, 0, stream>>>(s, d, n);
    };
    cvt(W_exp,  wexp_bf,  768 * 1536);
    cvt(cw,     cw_bf,    1536 * 512);
    cvt(W_proj, wproj_bf, 192 * 768);
    cvt(Wq,     wq_bf,    768 * 384);
    cvt(Wkv,    wkv_bf,   768 * 768);
    cvt(Wp2,    wp2_bf,   384 * 768);

    // 1) fea = x @ W_exp + b_exp  -> bf16
    gemm_wmma_kernel<float, bf16_t><<<dim3(M / 64, 1536 / 128), 256, 0, stream>>>(
        x, wexp_bf, b_exp, fea_bf, M, 1536, 768);

    // 2) ga = sigmoid(fea @ W_ga + b_ga)
    ga_kernel<<<(M * 8) / 256, 256, 0, stream>>>(fea_bf, W_ga, b_ga, ga);

    // 3) act = fea @ cluster_weights  -> f32
    gemm_wmma_kernel<bf16_t, float><<<dim3(M / 64, 512 / 128), 256, 0, stream>>>(
        fea_bf, cw_bf, (const float*)nullptr, act_f, M, 512, 1536);

    // 4) BN1 + softmax(C) * ga -> bf16
    act_softmax_kernel<<<(M * 8) / 256, 256, 0, stream>>>(
        act_f, bn1_g, bn1_b, bn1_m, bn1_v, ga, act_soft);

    // 5) cent = einsum('bnc,bnf->bcf')
    cent_wmma_kernel<<<32, 256, 0, stream>>>(act_soft, fea_bf, cent);

    // 6) nc = cent @ W_proj + b_proj  -> f32
    gemm_wmma_kernel<float, float><<<dim3(BC / 64, 768 / 128), 256, 0, stream>>>(
        cent, wproj_bf, b_proj, nc_f, BC, 768, 192);

    // 7) BN2 -> bf16
    bn2_kernel<<<(BC * 768 + 255) / 256, 256, 0, stream>>>(
        nc_f, bn2_g, bn2_b, bn2_m, bn2_v, nc_bf, BC * 768);

    // 8) q = x @ Wq + bq -> bf16
    gemm_wmma_kernel<float, bf16_t><<<dim3(M / 64, 384 / 128), 256, 0, stream>>>(
        x, wq_bf, bq, q_bf, M, 384, 768);

    // 9) kv = nc @ Wkv + bkv -> bf16
    gemm_wmma_kernel<bf16_t, bf16_t><<<dim3(BC / 64, 768 / 128), 256, 0, stream>>>(
        nc_bf, wkv_bf, bkv, kv_bf, BC, 768, 768);

    // 10) fused attention (TDM-staged KV) -> bf16
    (void)hipFuncSetAttribute((const void*)attn_kernel,
                              hipFuncAttributeMaxDynamicSharedMemorySize,
                              ATTN_SMEM_BYTES);
    attn_kernel<<<32 * 8, 256, ATTN_SMEM_BYTES, stream>>>(q_bf, kv_bf, ao_bf);

    // 11) out = attn_out @ Wp2 + bp2 -> f32
    gemm_wmma_kernel<bf16_t, float><<<dim3(M / 64, 768 / 128), 256, 0, stream>>>(
        ao_bf, wp2_bf, bp2, out, M, 768, 384);
}
